// ScaledDotProductAttention_40879498729023
// MI455X (gfx1250) — matrix operand order repaired and checked by reading, not compiled, not run on hardware
//
#include <hip/hip_runtime.h>

// Problem constants (from reference): B=16, LQ=2048, LK=2048, D=128
#define BB   16
#define LQ   2048
#define LK   2048
#define DD   128

typedef __attribute__((ext_vector_type(16))) _Float16 v16h;
typedef __attribute__((ext_vector_type(8)))  float    v8f;

union frag16 { v16h v; _Float16 e[16]; };
union acc8   { v8f  v; float    e[8];  };
union h8pack { _Float16 e[8]; float4 f; };   // force 16B LDS stores of f16 data

// ---- dynamic LDS layout (bytes) ----
// sS  : float   [16][2048]            @ 0       (131072 B)
// sQ  : _Float16[16][128]             @ 131072  (  4096 B)
// sV  : _Float16[32][136]  (padded)   @ 135168  (  8704 B)
// sRed: float   [16][16]              @ 143872  (  1024 B)
#define SMEM_S_OFF   0
#define SMEM_Q_OFF   131072
#define SMEM_V_OFF   135168
#define SMEM_R_OFF   143872
#define SMEM_BYTES   144896
#define VPITCH       136   // halves per staged V row: 272B = 16B-aligned, 4-bank k-stride

__global__ void attn_fwd_wmma_f16(const float* __restrict__ q,
                                  const float* __restrict__ k,
                                  const float* __restrict__ v,
                                  const float* __restrict__ scale_p,
                                  float* __restrict__ out_ctx,
                                  float* __restrict__ out_att)
{
    extern __shared__ char smem[];
    float*    sS   = (float*)(smem + SMEM_S_OFF);      // [16][2048] scores -> probs
    _Float16* sQ   = (_Float16*)(smem + SMEM_Q_OFF);   // [16][128]  Q tile f16
    _Float16* sV   = (_Float16*)(smem + SMEM_V_OFF);   // [32][VPITCH] V tile f16
    float*    sRed = (float*)(smem + SMEM_R_OFF);      // [16][16] reduction scratch

    const int tid  = threadIdx.x;          // 0..255 (8 wave32)
    const int lane = tid & 31;
    const int wave = tid >> 5;             // 0..7
    const int hi   = (lane >> 4) & 1;      // half-wave select
    const int ln   = lane & 15;

    const int b  = blockIdx.x / (LQ / 16);
    const int q0 = (blockIdx.x % (LQ / 16)) * 16;
    const float scale = scale_p[0];

    const float* qb = q + (size_t)b * LQ * DD + (size_t)q0 * DD;
    const float* kb = k + (size_t)b * LK * DD;
    const float* vb = v + (size_t)b * LK * DD;
    float* attb = out_att + (size_t)b * LQ * LK + (size_t)q0 * LK;
    float* ctxb = out_ctx + (size_t)b * LQ * DD + (size_t)q0 * DD;

    // ------------------------------------------------------------------
    // Phase 1: stage Q tile (16x128 f32 -> f16 LDS). 2048 elems, 8/thread.
    // ------------------------------------------------------------------
    {
        const int base = tid * 8;                 // covers 16*128 exactly
        const int row  = base >> 7;
        const int col  = base & 127;
        const float4* src = (const float4*)(qb + row * DD + col);
        float4 f0 = src[0], f1 = src[1];
        h8pack p;
        p.e[0] = (_Float16)f0.x; p.e[1] = (_Float16)f0.y;
        p.e[2] = (_Float16)f0.z; p.e[3] = (_Float16)f0.w;
        p.e[4] = (_Float16)f1.x; p.e[5] = (_Float16)f1.y;
        p.e[6] = (_Float16)f1.z; p.e[7] = (_Float16)f1.w;
        *(float4*)(sQ + row * 128 + col) = p.f;   // single ds_store_b128
    }
    __syncthreads();

    // ------------------------------------------------------------------
    // Phase 2: S = scale * Q K^T, WMMA f16 -> f32, scores into LDS.
    // A layout: row M = lane&15; half i -> Klocal = i + 16*hi
    // ------------------------------------------------------------------
    frag16 afrag[4];
#pragma unroll
    for (int ks = 0; ks < 4; ++ks) {
#pragma unroll
        for (int i = 0; i < 16; ++i) {
            const int kl = i + hi * 16;
            afrag[ks].e[i] = sQ[ln * 128 + ks * 32 + kl];
        }
    }

    for (int nt = wave; nt < LK / 16; nt += 8) {
        const int n = nt * 16 + ln;                       // key row handled by this lane
        const float* krow = kb + (size_t)n * DD;
        acc8 c; c.v = (v8f){};
#pragma unroll
        for (int ks = 0; ks < 4; ++ks) {
            // B layout: col N = lane&15; half i -> Klocal = i + 16*hi
            const float4* src = (const float4*)(krow + ks * 32 + hi * 16);
            float4 f0 = src[0], f1 = src[1], f2 = src[2], f3 = src[3];
            frag16 bf;
            bf.e[0]=(_Float16)f0.x; bf.e[1]=(_Float16)f0.y; bf.e[2]=(_Float16)f0.z; bf.e[3]=(_Float16)f0.w;
            bf.e[4]=(_Float16)f1.x; bf.e[5]=(_Float16)f1.y; bf.e[6]=(_Float16)f1.z; bf.e[7]=(_Float16)f1.w;
            bf.e[8]=(_Float16)f2.x; bf.e[9]=(_Float16)f2.y; bf.e[10]=(_Float16)f2.z; bf.e[11]=(_Float16)f2.w;
            bf.e[12]=(_Float16)f3.x; bf.e[13]=(_Float16)f3.y; bf.e[14]=(_Float16)f3.z; bf.e[15]=(_Float16)f3.w;
            c.v = __builtin_amdgcn_wmma_f32_16x16x32_f16(
                false, afrag[ks].v, false, bf.v, (short)0, c.v, false, false);
        }
        // C layout: VGPR r holds row (r + 8*hi), col = lane&15
#pragma unroll
        for (int r = 0; r < 8; ++r) {
            const int row = r + hi * 8;
            sS[row * LK + nt * 16 + ln] = c.e[r] * scale;
        }
    }
    __syncthreads();

    // ------------------------------------------------------------------
    // Early V prefetch: issue tile-0 global loads now so their latency is
    // hidden under the entire softmax phase.
    // ------------------------------------------------------------------
    const int vrow = tid >> 3;                     // 0..31
    const int vcol = (tid & 7) * 16;               // 0..112
    float4 vreg[4];
    {
        const float4* src = (const float4*)(vb + (size_t)vrow * DD + vcol);
#pragma unroll
        for (int jj = 0; jj < 4; ++jj) vreg[jj] = src[jj];
    }

    // ------------------------------------------------------------------
    // Phase 3: row softmax (16 rows x 16 reducer slots), float4 everywhere.
    // Thread (rrow, slot) owns float4 chunks at columns 4*(slot + 16*j).
    // ------------------------------------------------------------------
    const int rrow = tid >> 4;          // 0..15
    const int slot = tid & 15;          // 0..15
    float4* srow4 = (float4*)(sS + rrow * LK);

    float mloc = -__builtin_huge_valf();
    for (int j = 0; j < LK / 64; ++j) {
        float4 f = srow4[slot + j * 16];
        mloc = fmaxf(mloc, fmaxf(fmaxf(f.x, f.y), fmaxf(f.z, f.w)));
    }
    sRed[rrow * 16 + slot] = mloc;
    __syncthreads();
    if (slot == 0) {
        float mm = sRed[rrow * 16];
#pragma unroll
        for (int i = 1; i < 16; ++i) mm = fmaxf(mm, sRed[rrow * 16 + i]);
        sRed[rrow * 16] = mm;
    }
    __syncthreads();
    const float rowmax = sRed[rrow * 16];
    __syncthreads();                               // before sRed is reused for sums

    float sloc = 0.f;
    for (int j = 0; j < LK / 64; ++j) {
        float4 f = srow4[slot + j * 16];
        f.x = __expf(f.x - rowmax); f.y = __expf(f.y - rowmax);
        f.z = __expf(f.z - rowmax); f.w = __expf(f.w - rowmax);
        sloc += (f.x + f.y) + (f.z + f.w);
        srow4[slot + j * 16] = f;
    }
    sRed[rrow * 16 + slot] = sloc;
    __syncthreads();
    if (slot == 0) {
        float ss = sRed[rrow * 16];
#pragma unroll
        for (int i = 1; i < 16; ++i) ss += sRed[rrow * 16 + i];
        sRed[rrow * 16] = ss;
    }
    __syncthreads();
    const float inv = 1.0f / sRed[rrow * 16];

    float4* arow4 = (float4*)(attb + (size_t)rrow * LK);
    for (int j = 0; j < LK / 64; ++j) {
        float4 f = srow4[slot + j * 16];
        f.x *= inv; f.y *= inv; f.z *= inv; f.w *= inv;
        srow4[slot + j * 16] = f;                  // keep normalized probs for P@V
        arow4[slot + j * 16] = f;                  // coalesced 16B global store
    }

    // ------------------------------------------------------------------
    // Phase 4: ctx = P V with register-double-buffered V tiles.
    // Each wave owns D-segment [wave*16, wave*16+16); 64 k-steps of 32 keys.
    // ------------------------------------------------------------------
    acc8 acc; acc.v = (v8f){};
    const int dseg = wave * 16;

    for (int kt = 0; kt < LK / 32; ++kt) {
        __syncthreads();                           // all waves done reading prev sV
        {
            // commit prefetched V tile to LDS as f16 (two 16B stores)
            _Float16* dst = sV + vrow * VPITCH + vcol;
            h8pack p0, p1;
            p0.e[0]=(_Float16)vreg[0].x; p0.e[1]=(_Float16)vreg[0].y;
            p0.e[2]=(_Float16)vreg[0].z; p0.e[3]=(_Float16)vreg[0].w;
            p0.e[4]=(_Float16)vreg[1].x; p0.e[5]=(_Float16)vreg[1].y;
            p0.e[6]=(_Float16)vreg[1].z; p0.e[7]=(_Float16)vreg[1].w;
            p1.e[0]=(_Float16)vreg[2].x; p1.e[1]=(_Float16)vreg[2].y;
            p1.e[2]=(_Float16)vreg[2].z; p1.e[3]=(_Float16)vreg[2].w;
            p1.e[4]=(_Float16)vreg[3].x; p1.e[5]=(_Float16)vreg[3].y;
            p1.e[6]=(_Float16)vreg[3].z; p1.e[7]=(_Float16)vreg[3].w;
            *(float4*)(dst)     = p0.f;
            *(float4*)(dst + 8) = p1.f;
        }
        __syncthreads();

        if (kt + 1 < LK / 32) {                    // prefetch next tile -> regs,
            const float4* src =                    // overlaps with frag build + WMMA
                (const float4*)(vb + (size_t)((kt + 1) * 32 + vrow) * DD + vcol);
#pragma unroll
            for (int jj = 0; jj < 4; ++jj) vreg[jj] = src[jj];
        }

        // A = P tile (f32 -> f16), two contiguous 8-float runs -> 4x b128 LDS loads
        frag16 a;
        {
            const float* sArow = sS + ln * LK + kt * 32;
            const float4* p0 = (const float4*)(sArow + hi * 8);
            const float4* p1 = (const float4*)(sArow + 16 + hi * 8);
            float4 a0 = p0[0], a1 = p0[1], a2 = p1[0], a3 = p1[1];
            a.e[0]=(_Float16)a0.x;  a.e[1]=(_Float16)a0.y;
            a.e[2]=(_Float16)a0.z;  a.e[3]=(_Float16)a0.w;
            a.e[4]=(_Float16)a1.x;  a.e[5]=(_Float16)a1.y;
            a.e[6]=(_Float16)a1.z;  a.e[7]=(_Float16)a1.w;
            a.e[8]=(_Float16)a2.x;  a.e[9]=(_Float16)a2.y;
            a.e[10]=(_Float16)a2.z; a.e[11]=(_Float16)a2.w;
            a.e[12]=(_Float16)a3.x; a.e[13]=(_Float16)a3.y;
            a.e[14]=(_Float16)a3.z; a.e[15]=(_Float16)a3.w;
        }
        // B = V tile column (dseg+ln); half i -> Krow = (i<8 ? i+8*hi : 16+(i-8)+8*hi)
        frag16 bf;
#pragma unroll
        for (int i = 0; i < 16; ++i) {
            bf.e[i] = sV[((i < 8) ? (i + hi * 8) : (16 + (i - 8) + hi * 8)) * VPITCH + dseg + ln];
        }
        acc.v = __builtin_amdgcn_wmma_f32_16x16x32_f16(
            false, a.v, false, bf.v, (short)0, acc.v, false, false);
    }

#pragma unroll
    for (int r = 0; r < 8; ++r) {
        const int row = r + hi * 8;
        ctxb[row * DD + dseg + ln] = acc.e[r];
    }
}

extern "C" void kernel_launch(void* const* d_in, const int* in_sizes, int n_in,
                              void* d_out, int out_size, void* d_ws, size_t ws_size,
                              hipStream_t stream) {
    (void)in_sizes; (void)n_in; (void)out_size; (void)d_ws; (void)ws_size;
    const float* q     = (const float*)d_in[0];
    const float* k     = (const float*)d_in[1];
    const float* v     = (const float*)d_in[2];
    const float* scale = (const float*)d_in[3];

    // d_out = [context (B*LQ*D) | attention (B*LQ*LK)] in return order
    float* out_ctx = (float*)d_out;
    float* out_att = (float*)d_out + (size_t)BB * LQ * DD;

    (void)hipFuncSetAttribute((const void*)attn_fwd_wmma_f16,
                              hipFuncAttributeMaxDynamicSharedMemorySize,
                              SMEM_BYTES);

    const int grid = BB * (LQ / 16);   // 2048 blocks, one 16-row Q tile each
    attn_fwd_wmma_f16<<<dim3(grid), dim3(256), SMEM_BYTES, stream>>>(
        q, k, v, scale, out_ctx, out_att);
}
